// Block_64192581206029
// MI455X (gfx1250) — compile-verified, matrix operand order checked by reading, not run on hardware
//
#include <hip/hip_runtime.h>
#include <hip/hip_bf16.h>

typedef __bf16 bf16_t;
typedef bf16_t v16bf __attribute__((ext_vector_type(16)));
typedef bf16_t v8bf  __attribute__((ext_vector_type(8)));
typedef float  v8f   __attribute__((ext_vector_type(8)));
typedef unsigned int u32;
typedef u32 v4u __attribute__((ext_vector_type(4)));
typedef int  v8i __attribute__((ext_vector_type(8)));
typedef int  v4i __attribute__((ext_vector_type(4)));

#define D_MODEL 1024
#define N_HEADS 16
#define HEAD_DIM 64
#define BB 2
#define TT 2048
#define NTOK (BB * TT)      /* 4096 tokens */
#define HID (4 * D_MODEL)   /* 4096 */

// GEMM LDS tile row stride (bf16 elems): 80B rows -> 16B aligned b128 loads AND
// conflict-free fragment reads (lane stride 20 dwords; 5d % 16 != 0 for d=1..15)
#define LDSR 40

#if __has_builtin(__builtin_amdgcn_tensor_load_to_lds)
#define HAVE_TDM 1
#endif

__device__ __forceinline__ v8f zero8() {
  v8f z = {0.f, 0.f, 0.f, 0.f, 0.f, 0.f, 0.f, 0.f};
  return z;
}

// ---- attention LDS swizzles (rows of 32 / 64 bf16) ----------------------
__device__ __forceinline__ int swzA(int row, int chunk) {
  return row * 32 + ((chunk ^ (row >> 2)) & 3) * 8;
}
__device__ __forceinline__ int swzK(int row, int chunk) {
  return row * 64 + ((chunk ^ (row >> 1)) & 7) * 8;
}

// A-layout operand fragment from swizzled LDS (attention)
__device__ __forceinline__ v16bf load_frag_swz(const bf16_t* base, int row, int h) {
  v8bf lo = *(const v8bf*)(base + swzA(row, h));
  v8bf hi = *(const v8bf*)(base + swzA(row, 2 + h));
  v16bf f;
#pragma unroll
  for (int i = 0; i < 8; ++i) { f[i] = lo[i]; f[8 + i] = hi[i]; }
  return f;
}

// A-layout operand fragment from padded-linear LDS (GEMM; matches TDM pad)
__device__ __forceinline__ v16bf load_frag_pad(const bf16_t* base, int row, int h) {
  v8bf lo = *(const v8bf*)(base + row * LDSR + h * 8);
  v8bf hi = *(const v8bf*)(base + row * LDSR + 16 + h * 8);
  v16bf f;
#pragma unroll
  for (int i = 0; i < 8; ++i) { f[i] = lo[i]; f[8 + i] = hi[i]; }
  return f;
}

#ifdef HAVE_TDM
// Issue one TDM 2D tile load: 128 rows x 32 bf16, row-major source with
// row stride = strideElems (bf16), padded in LDS to LDSR-elem rows via the
// descriptor pad feature (pad_interval=16 dwords -> code 3, pad_amount=4
// dwords -> code 3). D# layout per CDNA5 ISA 8.3/8.4.
// Toolchain builtin arity: 6 args (g0 x4, g1 x8, g2 x4, g3 x4, extra x8, cpol).
__device__ __forceinline__ void tdm_load_tile_128x32(u32 lds_off,
                                                     const bf16_t* tile_ptr,
                                                     u32 dim0Elems, u32 dim1Rows,
                                                     u32 strideElems) {
  unsigned long long ga = (unsigned long long)(size_t)tile_ptr;
  v4u g0;
  g0[0] = 1u;                                   // count=1, user descriptor
  g0[1] = lds_off;                              // lds_addr
  g0[2] = (u32)ga;                              // global_addr[31:0]
  g0[3] = (u32)((ga >> 32) & 0x01FFFFFFu) | (2u << 30);  // addr[56:32], type=2
  v8i g1;
  g1[0] = (int)((1u << 16)      // data_size = 2B
                | (1u << 20)    // pad_enable
                | (3u << 22)    // pad_interval: 16 dwords per pad
                | (3u << 25));  // pad_amount: 4 dwords
  g1[1] = (int)((dim0Elems & 0xFFFFu) << 16);                       // tensor_dim0 lo
  g1[2] = (int)((dim0Elems >> 16) | ((dim1Rows & 0xFFFFu) << 16));  // d0 hi | d1 lo
  g1[3] = (int)((dim1Rows >> 16) | (32u << 16));                    // d1 hi | tile_dim0=32
  g1[4] = 128;                                                      // tile_dim1=128
  g1[5] = (int)strideElems;                                         // dim0_stride lo
  g1[6] = 0;
  g1[7] = 0;
  v4i zz4 = {0, 0, 0, 0};
  v8i zz8 = {0, 0, 0, 0, 0, 0, 0, 0};
  __builtin_amdgcn_tensor_load_to_lds(g0, g1, zz4, zz4, zz8, 0);
}

__device__ __forceinline__ void wait_tensor0() {
#if __has_builtin(__builtin_amdgcn_s_wait_tensorcnt)
  __builtin_amdgcn_s_wait_tensorcnt(0);
#else
  asm volatile("s_wait_tensorcnt 0" ::: "memory");
#endif
}
#endif  // HAVE_TDM

// ---- fused fp32->bf16 transpose: in [K][N] f32 -> out [N][K] bf16 -------
__global__ __launch_bounds__(256) void transpose_cvt(const float* __restrict__ in,
                                                     bf16_t* __restrict__ out,
                                                     int K, int N) {
  __shared__ float t[64][65];
  const int kb = blockIdx.y * 64, nb = blockIdx.x * 64;
  const int tx = threadIdx.x & 63, ty = threadIdx.x >> 6;
#pragma unroll
  for (int i = 0; i < 64; i += 4)
    t[ty + i][tx] = in[(size_t)(kb + ty + i) * N + nb + tx];
  __syncthreads();
#pragma unroll
  for (int i = 0; i < 64; i += 4)
    out[(size_t)(nb + ty + i) * K + kb + tx] = (bf16_t)t[tx][ty + i];
}

// ---- LayerNorm: one block per token row, bf16 output --------------------
__global__ __launch_bounds__(256) void layernorm_bf16(const float* __restrict__ x,
                                                      const float* __restrict__ g,
                                                      const float* __restrict__ bta,
                                                      bf16_t* __restrict__ out) {
  __shared__ float red[256];
  const int row = blockIdx.x;
  const int tid = threadIdx.x;
  const float* xr = x + (size_t)row * D_MODEL;
  float v[4];
  float s = 0.f;
#pragma unroll
  for (int i = 0; i < 4; ++i) { v[i] = xr[tid + i * 256]; s += v[i]; }
  red[tid] = s;
  __syncthreads();
  for (int o = 128; o > 0; o >>= 1) { if (tid < o) red[tid] += red[tid + o]; __syncthreads(); }
  const float mu = red[0] * (1.f / D_MODEL);
  __syncthreads();
  float s2 = 0.f;
#pragma unroll
  for (int i = 0; i < 4; ++i) { float d = v[i] - mu; s2 += d * d; }
  red[tid] = s2;
  __syncthreads();
  for (int o = 128; o > 0; o >>= 1) { if (tid < o) red[tid] += red[tid + o]; __syncthreads(); }
  const float rstd = rsqrtf(red[0] * (1.f / D_MODEL) + 1e-5f);
  bf16_t* orow = out + (size_t)row * D_MODEL;
#pragma unroll
  for (int i = 0; i < 4; ++i) {
    int c = tid + i * 256;
    orow[c] = (bf16_t)((v[i] - mu) * rstd * g[c] + bta[c]);
  }
}

// ---- bf16 WMMA GEMM: C[M,N] = A[M,K] @ Bt[N,K]^T + bias -----------------
// mode 0: store bf16 ; mode 1: exact GELU, bf16 ; mode 2: +resid, f32
__global__ __launch_bounds__(256) void gemm_bf16_wmma(
    const bf16_t* __restrict__ A, const bf16_t* __restrict__ Bt,
    const float* __restrict__ bias, const float* __restrict__ resid,
    bf16_t* __restrict__ outB, float* __restrict__ outF,
    int M, int N, int K, int mode) {
  __shared__ bf16_t sm[2][2][128 * LDSR];  // [buf][A/B][tile]
  const int tid = threadIdx.x;
  const int lane = tid & 31, wid = tid >> 5;
  const int r = lane & 15, h = lane >> 4;
  const int waveM = (wid & 1) * 64, waveN = (wid >> 1) * 32;
  const int m0 = blockIdx.y * 128, n0 = blockIdx.x * 128;

  v8f acc[4][2];
#pragma unroll
  for (int a = 0; a < 4; ++a)
#pragma unroll
    for (int b = 0; b < 2; ++b) acc[a][b] = zero8();

#ifdef HAVE_TDM
  // -------- TDM double-buffered pipeline --------
  if (wid == 0) {
    tdm_load_tile_128x32((u32)(size_t)&sm[0][0][0], A + (size_t)m0 * K, K, M, K);
    tdm_load_tile_128x32((u32)(size_t)&sm[0][1][0], Bt + (size_t)n0 * K, K, N, K);
  }
  int cur = 0;
  for (int k0 = 0; k0 < K; k0 += 32) {
    wait_tensor0();       // no-op for non-issuing waves (TENSORcnt==0)
    __syncthreads();      // tile `cur` visible; all waves done with `cur^1`
    if (wid == 0 && k0 + 32 < K) {
      tdm_load_tile_128x32((u32)(size_t)&sm[cur ^ 1][0][0],
                           A + (size_t)m0 * K + (k0 + 32), K, M, K);
      tdm_load_tile_128x32((u32)(size_t)&sm[cur ^ 1][1][0],
                           Bt + (size_t)n0 * K + (k0 + 32), K, N, K);
    }
    const bf16_t* As = sm[cur][0];
    const bf16_t* Bs = sm[cur][1];
    v16bf af[4], bfr[2];
#pragma unroll
    for (int mt = 0; mt < 4; ++mt) af[mt] = load_frag_pad(As, waveM + mt * 16 + r, h);
#pragma unroll
    for (int nt = 0; nt < 2; ++nt) bfr[nt] = load_frag_pad(Bs, waveN + nt * 16 + r, h);
#pragma unroll
    for (int mt = 0; mt < 4; ++mt)
#pragma unroll
      for (int nt = 0; nt < 2; ++nt)
        acc[mt][nt] = __builtin_amdgcn_wmma_f32_16x16x32_bf16(
            false, af[mt], false, bfr[nt], (short)0, acc[mt][nt], false, false);
    cur ^= 1;
  }
#else
  // -------- fallback: cooperative vector loads --------
  (void)M;
  bf16_t* As = sm[0][0];
  bf16_t* Bs = sm[0][1];
  const int rr = tid >> 2, c = tid & 3;
  for (int k0 = 0; k0 < K; k0 += 32) {
#pragma unroll
    for (int it = 0; it < 2; ++it) {
      int row = rr + it * 64;
      *(v8bf*)(As + row * LDSR + c * 8) =
          *(const v8bf*)(A + (size_t)(m0 + row) * K + k0 + c * 8);
      *(v8bf*)(Bs + row * LDSR + c * 8) =
          *(const v8bf*)(Bt + (size_t)(n0 + row) * K + k0 + c * 8);
    }
    __syncthreads();
    v16bf af[4], bfr[2];
#pragma unroll
    for (int mt = 0; mt < 4; ++mt) af[mt] = load_frag_pad(As, waveM + mt * 16 + r, h);
#pragma unroll
    for (int nt = 0; nt < 2; ++nt) bfr[nt] = load_frag_pad(Bs, waveN + nt * 16 + r, h);
#pragma unroll
    for (int mt = 0; mt < 4; ++mt)
#pragma unroll
      for (int nt = 0; nt < 2; ++nt)
        acc[mt][nt] = __builtin_amdgcn_wmma_f32_16x16x32_bf16(
            false, af[mt], false, bfr[nt], (short)0, acc[mt][nt], false, false);
    __syncthreads();
  }
#endif

#pragma unroll
  for (int mt = 0; mt < 4; ++mt) {
#pragma unroll
    for (int nt = 0; nt < 2; ++nt) {
      const int col = n0 + waveN + nt * 16 + r;
      const float bcol = bias[col];
#pragma unroll
      for (int i = 0; i < 8; ++i) {
        const int row = m0 + waveM + mt * 16 + i + 8 * h;
        float v = acc[mt][nt][i] + bcol;
        if (mode == 1) v = 0.5f * v * (1.0f + erff(v * 0.70710678118654752f));
        if (mode == 2) {
          v += resid[(size_t)row * N + col];
          outF[(size_t)row * N + col] = v;
        } else {
          outB[(size_t)row * N + col] = (bf16_t)v;
        }
      }
    }
  }
}

// ---- Flash attention (causal), bf16 WMMA, f32 online softmax ------------
__global__ __launch_bounds__(256) void attn_wmma(const bf16_t* __restrict__ qkv,
                                                 bf16_t* __restrict__ out) {
  __shared__ bf16_t Ks[32 * 64];
  __shared__ bf16_t Vt[64 * 32];
  __shared__ bf16_t Ps[8 * 16 * 32];
  const int tid = threadIdx.x;
  const int lane = tid & 31, wid = tid >> 5;
  const int r = lane & 15, h = lane >> 4;
  const int qBase = blockIdx.x * 128;
  const int head = blockIdx.y;
  const int bIdx = blockIdx.z;
  const size_t rowStride = 3 * D_MODEL;
  const bf16_t* qk = qkv + (size_t)bIdx * TT * rowStride;

  const int qRow0 = qBase + wid * 16;
  v16bf qf[2];
  {
    const bf16_t* qrow = qk + (size_t)(qRow0 + r) * rowStride + head * HEAD_DIM;
#pragma unroll
    for (int kd = 0; kd < 2; ++kd) {
      v8bf lo = *(const v8bf*)(qrow + kd * 32 + h * 8);
      v8bf hi = *(const v8bf*)(qrow + kd * 32 + 16 + h * 8);
#pragma unroll
      for (int i = 0; i < 8; ++i) { qf[kd][i] = lo[i]; qf[kd][8 + i] = hi[i]; }
    }
  }

  v8f O[4];
#pragma unroll
  for (int nt = 0; nt < 4; ++nt) O[nt] = zero8();
  float mrow[8], lrow[8];
#pragma unroll
  for (int i = 0; i < 8; ++i) { mrow[i] = -__builtin_inff(); lrow[i] = 0.f; }

  const int cKey = tid >> 3, cChunk = tid & 7;
  const int nTiles = qBase / 32 + 4;
  bf16_t* myPs = Ps + wid * 16 * 32;

  for (int kt = 0; kt < nTiles; ++kt) {
    {
      const bf16_t* krow =
          qk + (size_t)(kt * 32 + cKey) * rowStride + D_MODEL + head * HEAD_DIM;
      v8bf kv = *(const v8bf*)(krow + cChunk * 8);
      *(v8bf*)(Ks + swzK(cKey, cChunk)) = kv;
      const bf16_t* vrow =
          qk + (size_t)(kt * 32 + cKey) * rowStride + 2 * D_MODEL + head * HEAD_DIM;
      v8bf vv = *(const v8bf*)(vrow + cChunk * 8);
#pragma unroll
      for (int j = 0; j < 8; ++j) {
        int dim = cChunk * 8 + j;
        Vt[swzA(dim, cKey >> 3) + (cKey & 7)] = vv[j];
      }
    }
    __syncthreads();

    v8f S[2];
#pragma unroll
    for (int nt = 0; nt < 2; ++nt) {
      S[nt] = zero8();
#pragma unroll
      for (int kd = 0; kd < 2; ++kd) {
        int krowi = nt * 16 + r;
        v8bf lo = *(const v8bf*)(Ks + swzK(krowi, kd * 4 + h));
        v8bf hi = *(const v8bf*)(Ks + swzK(krowi, kd * 4 + 2 + h));
        v16bf kf;
#pragma unroll
        for (int i = 0; i < 8; ++i) { kf[i] = lo[i]; kf[8 + i] = hi[i]; }
        S[nt] = __builtin_amdgcn_wmma_f32_16x16x32_bf16(
            false, qf[kd], false, kf, (short)0, S[nt], false, false);
      }
    }

#pragma unroll
    for (int i = 0; i < 8; ++i) {
      const int qAbs = qRow0 + i + 8 * h;
      float s0 = S[0][i] * 0.125f;
      float s1 = S[1][i] * 0.125f;
      if (kt * 32 + r > qAbs) s0 = -__builtin_inff();
      if (kt * 32 + 16 + r > qAbs) s1 = -__builtin_inff();
      float mc = fmaxf(s0, s1);
#pragma unroll
      for (int off = 1; off < 16; off <<= 1) mc = fmaxf(mc, __shfl_xor(mc, off, 32));
      const float mnew = fmaxf(mrow[i], mc);
      const float alpha = __expf(mrow[i] - mnew);
      const float p0 = __expf(s0 - mnew);
      const float p1 = __expf(s1 - mnew);
      float ps = p0 + p1;
#pragma unroll
      for (int off = 1; off < 16; off <<= 1) ps += __shfl_xor(ps, off, 32);
      lrow[i] = lrow[i] * alpha + ps;
      mrow[i] = mnew;
#pragma unroll
      for (int nt = 0; nt < 4; ++nt) O[nt][i] *= alpha;
      const int prow = i + 8 * h;
      myPs[swzA(prow, r >> 3) + (r & 7)] = (bf16_t)p0;
      myPs[swzA(prow, 2 + (r >> 3)) + (r & 7)] = (bf16_t)p1;
    }
    asm volatile("s_wait_dscnt 0" ::: "memory");

    v16bf pf = load_frag_swz(myPs, r, h);
#pragma unroll
    for (int nt = 0; nt < 4; ++nt) {
      int drow = nt * 16 + r;
      v8bf lo = *(const v8bf*)(Vt + swzA(drow, h));
      v8bf hi = *(const v8bf*)(Vt + swzA(drow, 2 + h));
      v16bf vf;
#pragma unroll
      for (int i = 0; i < 8; ++i) { vf[i] = lo[i]; vf[8 + i] = hi[i]; }
      O[nt] = __builtin_amdgcn_wmma_f32_16x16x32_bf16(
          false, pf, false, vf, (short)0, O[nt], false, false);
    }
    __syncthreads();
  }

#pragma unroll
  for (int i = 0; i < 8; ++i) {
    const float inv = 1.f / lrow[i];
    const int t = qRow0 + i + 8 * h;
#pragma unroll
    for (int nt = 0; nt < 4; ++nt) {
      out[(size_t)(bIdx * TT + t) * D_MODEL + head * HEAD_DIM + nt * 16 + r] =
          (bf16_t)(O[nt][i] * inv);
    }
  }
}

// ---- host-side orchestration -------------------------------------------
extern "C" void kernel_launch(void* const* d_in, const int* in_sizes, int n_in,
                              void* d_out, int out_size, void* d_ws, size_t ws_size,
                              hipStream_t stream) {
  (void)in_sizes; (void)n_in; (void)out_size; (void)ws_size;
  const float* x      = (const float*)d_in[0];
  const float* ln1_g  = (const float*)d_in[2];
  const float* ln1_b  = (const float*)d_in[3];
  const float* qkv_w  = (const float*)d_in[4];
  const float* qkv_b  = (const float*)d_in[5];
  const float* proj_w = (const float*)d_in[6];
  const float* proj_b = (const float*)d_in[7];
  const float* ln2_g  = (const float*)d_in[8];
  const float* ln2_b  = (const float*)d_in[9];
  const float* w1     = (const float*)d_in[10];
  const float* b1     = (const float*)d_in[11];
  const float* w2     = (const float*)d_in[12];
  const float* b2     = (const float*)d_in[13];
  float* outp = (float*)d_out;

  char* ws = (char*)d_ws;
  size_t off = 0;
  auto alloc = [&](size_t bytes) -> void* {
    void* p = ws + off;
    off = (off + bytes + 255) & ~(size_t)255;
    return p;
  };
  bf16_t* qkvw_t = (bf16_t*)alloc((size_t)3 * D_MODEL * D_MODEL * 2);  // [3d][d]
  bf16_t* projw_t = (bf16_t*)alloc((size_t)D_MODEL * D_MODEL * 2);     // [d][d]
  bf16_t* w1_t = (bf16_t*)alloc((size_t)HID * D_MODEL * 2);            // [h][d]
  bf16_t* w2_t = (bf16_t*)alloc((size_t)D_MODEL * HID * 2);            // [d][h]
  bf16_t* xhat = (bf16_t*)alloc((size_t)NTOK * D_MODEL * 2);
  bf16_t* qkv_a = (bf16_t*)alloc((size_t)NTOK * 3 * D_MODEL * 2);
  bf16_t* attout = (bf16_t*)alloc((size_t)NTOK * D_MODEL * 2);
  float* x1 = (float*)alloc((size_t)NTOK * D_MODEL * 4);
  bf16_t* hhat = (bf16_t*)alloc((size_t)NTOK * D_MODEL * 2);
  bf16_t* h_b = (bf16_t*)alloc((size_t)NTOK * HID * 2);

  // one-time weight convert+transpose: [K][N] f32 -> [N][K] bf16
  transpose_cvt<<<dim3(3 * D_MODEL / 64, D_MODEL / 64), 256, 0, stream>>>(
      qkv_w, qkvw_t, D_MODEL, 3 * D_MODEL);
  transpose_cvt<<<dim3(D_MODEL / 64, D_MODEL / 64), 256, 0, stream>>>(
      proj_w, projw_t, D_MODEL, D_MODEL);
  transpose_cvt<<<dim3(HID / 64, D_MODEL / 64), 256, 0, stream>>>(
      w1, w1_t, D_MODEL, HID);
  transpose_cvt<<<dim3(D_MODEL / 64, HID / 64), 256, 0, stream>>>(
      w2, w2_t, HID, D_MODEL);

  // ln1 -> xhat (bf16)
  layernorm_bf16<<<NTOK, 256, 0, stream>>>(x, ln1_g, ln1_b, xhat);

  // qkv = xhat @ qkv_w + qkv_b
  gemm_bf16_wmma<<<dim3(3 * D_MODEL / 128, NTOK / 128), 256, 0, stream>>>(
      xhat, qkvw_t, qkv_b, nullptr, qkv_a, nullptr, NTOK, 3 * D_MODEL, D_MODEL, 0);

  // causal flash attention
  attn_wmma<<<dim3(TT / 128, N_HEADS, BB), 256, 0, stream>>>(qkv_a, attout);

  // x1 = x + attout @ proj_w + proj_b
  gemm_bf16_wmma<<<dim3(D_MODEL / 128, NTOK / 128), 256, 0, stream>>>(
      attout, projw_t, proj_b, x, nullptr, x1, NTOK, D_MODEL, D_MODEL, 2);

  // ln2 -> hhat (bf16)
  layernorm_bf16<<<NTOK, 256, 0, stream>>>(x1, ln2_g, ln2_b, hhat);

  // h = gelu(hhat @ w1 + b1)
  gemm_bf16_wmma<<<dim3(HID / 128, NTOK / 128), 256, 0, stream>>>(
      hhat, w1_t, b1, nullptr, h_b, nullptr, NTOK, HID, D_MODEL, 1);

  // out = x1 + h @ w2 + b2
  gemm_bf16_wmma<<<dim3(D_MODEL / 128, NTOK / 128), 256, 0, stream>>>(
      h_b, w2_t, b2, x1, nullptr, outp, NTOK, D_MODEL, HID, 2);
}
